// ScaledDotProductAttention_35201551958770
// MI455X (gfx1250) — matrix operand order repaired and checked by reading, not compiled, not run on hardware
//
#include <hip/hip_runtime.h>

typedef __bf16 bf16_t;
typedef __attribute__((ext_vector_type(16))) __bf16 v16bf;
typedef __attribute__((ext_vector_type(8)))  __bf16 v8bf;
typedef __attribute__((ext_vector_type(8)))  float   v8f;

#define S_LEN 2048
#define D_DIM 64
#define N_BH  32          // B*H = 2*16
#define N_HEADS 16
#define QT    32          // query rows per workgroup
#define SPAD  2056        // padded LDS row stride in bf16 elems (2048 + 8 -> bank-conflict-free)
#define SW_BYTES (QT * SPAD * 2)
#define SMEM_BYTES (SW_BYTES + 256*4 + 256*4 + 32*4 + 32*4)

static __device__ __forceinline__ v8f wmma_bf16(v16bf a, v16bf b, v8f c) {
  // v_wmma_f32_16x16x32_bf16: (neg_a, A, neg_b, B, c_mod, C, reuse_a, reuse_b)
  return __builtin_amdgcn_wmma_f32_16x16x32_bf16(false, a, false, b, (short)0, c,
                                                 false, false);
}

static __device__ __forceinline__ void cvt8(float4 a, float4 b, v16bf& d, int base) {
  d[base + 0] = (bf16_t)a.x; d[base + 1] = (bf16_t)a.y;
  d[base + 2] = (bf16_t)a.z; d[base + 3] = (bf16_t)a.w;
  d[base + 4] = (bf16_t)b.x; d[base + 5] = (bf16_t)b.y;
  d[base + 6] = (bf16_t)b.z; d[base + 7] = (bf16_t)b.w;
}

// ---------------------------------------------------------------------------
// Pre-pass: Vt[bh][d][s] = (bf16) V[bh][s][d]  — gives contiguous-k B-fragments
// for the P*V WMMA. LDS-tiled 64x64 transpose, coalesced both directions.
// ---------------------------------------------------------------------------
__global__ __launch_bounds__(256) void vtranspose_kernel(const float* __restrict__ V,
                                                         bf16_t* __restrict__ Vt) {
  __shared__ float tile[64][65];
  const int bh = blockIdx.x;
  const int s0 = blockIdx.y * 64;
  const int t  = threadIdx.x;
  const int c  = t & 63;        // inner (coalesced) index
  const int r4 = t >> 6;        // 0..3

  const float* vp = V + ((size_t)bh * S_LEN + s0) * D_DIM;
#pragma unroll
  for (int rr = r4; rr < 64; rr += 4)
    tile[c][rr] = vp[(size_t)rr * D_DIM + c];   // read row rr (64 floats, coalesced)
  __syncthreads();

  bf16_t* op = Vt + (size_t)bh * D_DIM * S_LEN + s0;
#pragma unroll
  for (int dd = r4; dd < 64; dd += 4)
    op[(size_t)dd * S_LEN + c] = (bf16_t)tile[dd][c];  // write d-row (coalesced bf16)
}

// ---------------------------------------------------------------------------
// Fused attention: one workgroup = 32 query rows of one (b,h).
//   Phase 1 : scores = mask(QK^T/8) via WMMA bf16 -> LDS (bf16)
//   Phase 1.5: per-row online (max, sum-exp)
//   Phase 2 : weights = exp(s-max)/sum -> HBM (f32) and back to LDS (bf16)
//   Phase 3 : context = W @ V via WMMA bf16 (A from LDS, B from Vt in d_ws)
// ---------------------------------------------------------------------------
__global__ __launch_bounds__(256) void attn_kernel(const float* __restrict__ Q,
                                                   const float* __restrict__ K,
                                                   const int*   __restrict__ mask,
                                                   const bf16_t* __restrict__ Vt,
                                                   float* __restrict__ ctx_out,
                                                   float* __restrict__ w_out) {
  extern __shared__ unsigned char smem[];
  bf16_t* sw     = (bf16_t*)smem;                       // QT x SPAD bf16 scores/weights
  float*  redm   = (float*)(smem + SW_BYTES);           // 256 per-thread partial max
  float*  reds   = redm + 256;                          // 256 per-thread partial sum
  float*  rowm   = reds + 256;                          // 32 row max
  float*  rowsum = rowm + 32;                           // 32 row sum

  const int bh = blockIdx.x >> 6;            // 32 heads
  const int q0 = (blockIdx.x & 63) * QT;     // 64 query tiles of 32 rows
  const int b  = bh >> 4;                    // batch (H = 16)

  const int t    = threadIdx.x;
  const int w    = t >> 5;                   // wave 0..7
  const int lane = t & 31;
  const int half = lane >> 4;                // lane half (0/1)
  const int nl   = lane & 15;

  const float* Qh    = Q + (size_t)bh * S_LEN * D_DIM;
  const float* Kh    = K + (size_t)bh * S_LEN * D_DIM;
  const int*   maskb = mask + (size_t)b * S_LEN * S_LEN;
  float*       wout  = w_out   + (size_t)bh * S_LEN * S_LEN;
  float*       cout  = ctx_out + (size_t)bh * S_LEN * D_DIM;

  const int r = w & 1;                       // 16-row subtile within the 32-row block

  // ---- Phase 1: QK^T -> masked/scaled scores in LDS --------------------------
  // A fragments (Q rows, loop-invariant): lane holds row m = nl;
  //   elems 0..7 -> k = kb + 8*half + i ; elems 8..15 -> k = kb + 16 + 8*half + i
  v16bf a0, a1;
  {
    const float* qp = Qh + (size_t)(q0 + 16 * r + nl) * D_DIM + 8 * half;
    const float4* q4 = (const float4*)qp;
    cvt8(q4[0], q4[1], a0, 0);   // k =  0..: [8h, 8h+8)
    cvt8(q4[4], q4[5], a0, 8);   // k = 16..: offset +16 floats = 4 float4
    cvt8(q4[8], q4[9], a1, 0);   // k = 32..
    cvt8(q4[12], q4[13], a1, 8); // k = 48..
  }

  for (int ct = (w >> 1); ct < S_LEN / 16; ct += 4) {
    // B fragment: column n = nl -> K row ct*16+nl; elems 0..7 -> k = kb + 8*half + i ; elems 8..15 -> k = kb + 16 + 8*half + i
    const float* kp = Kh + (size_t)(ct * 16 + nl) * D_DIM + 8 * half;
    const float4* k4 = (const float4*)kp;
    v16bf b0, b1;
    cvt8(k4[0], k4[1], b0, 0);
    cvt8(k4[4], k4[5], b0, 8);
    cvt8(k4[8], k4[9], b1, 0);   // +32 floats
    cvt8(k4[12], k4[13], b1, 8);

    v8f c = {0.f, 0.f, 0.f, 0.f, 0.f, 0.f, 0.f, 0.f};
    c = wmma_bf16(a0, b0, c);
    c = wmma_bf16(a1, b1, c);

    // C layout: vgpr v -> row m = v + 8*half, col n = nl
    const int coln = ct * 16 + nl;
    const int rowb = 16 * r + 8 * half;
    const int* mp = maskb + (size_t)(q0 + rowb) * S_LEN + coln;
    bf16_t* sp = sw + (size_t)rowb * SPAD + coln;
#pragma unroll
    for (int v = 0; v < 8; ++v) {
      float s = c[v] * 0.125f;                 // 1/sqrt(64)
      if (mp[(size_t)v * S_LEN] == 0) s = -1e9f;
      sp[(size_t)v * SPAD] = (bf16_t)s;
    }
  }
  __syncthreads();

  // ---- Phase 1.5: per-row max & sum(exp) ------------------------------------
  const int row = t >> 3;                      // 8 threads per row
  const int sub = t & 7;
  {
    float mx = -3.0e38f, sm = 0.f;
    const v8bf* rp = (const v8bf*)(sw + (size_t)row * SPAD);
    for (int c8 = sub; c8 < S_LEN / 8; c8 += 8) {
      v8bf x = rp[c8];
      float f[8];
#pragma unroll
      for (int i = 0; i < 8; ++i) f[i] = (float)x[i];
      float cm = f[0];
#pragma unroll
      for (int i = 1; i < 8; ++i) cm = fmaxf(cm, f[i]);
      const float nm = fmaxf(mx, cm);
      float acc = 0.f;
#pragma unroll
      for (int i = 0; i < 8; ++i) acc += __expf(f[i] - nm);
      sm = sm * __expf(mx - nm) + acc;
      mx = nm;
    }
    redm[t] = mx; reds[t] = sm;
  }
  __syncthreads();
  if (sub == 0) {
    float M = redm[t], Ssum = reds[t];
#pragma unroll
    for (int i = 1; i < 8; ++i) {
      const float m2 = redm[t + i], s2 = reds[t + i];
      const float nm = fmaxf(M, m2);
      Ssum = Ssum * __expf(M - nm) + s2 * __expf(m2 - nm);
      M = nm;
    }
    rowm[row] = M; rowsum[row] = Ssum;
  }
  __syncthreads();

  // ---- Phase 2: normalize -> HBM weights (f32) + LDS weights (bf16) ---------
  {
    const float M = rowm[row];
    const float inv = 1.0f / rowsum[row];
    v8bf* rp = (v8bf*)(sw + (size_t)row * SPAD);
    float* gw = wout + (size_t)(q0 + row) * S_LEN;
    for (int c8 = sub; c8 < S_LEN / 8; c8 += 8) {
      v8bf x = rp[c8];
      float wv[8];
      v8bf y;
#pragma unroll
      for (int i = 0; i < 8; ++i) {
        wv[i] = __expf((float)x[i] - M) * inv;
        y[i] = (bf16_t)wv[i];
      }
      float4* g4 = (float4*)(gw + c8 * 8);
      g4[0] = make_float4(wv[0], wv[1], wv[2], wv[3]);
      g4[1] = make_float4(wv[4], wv[5], wv[6], wv[7]);
      rp[c8] = y;                               // in-place bf16 weights for P*V
    }
  }
  __syncthreads();

  // ---- Phase 3: context = W @ V (A = LDS weights, B = Vt[bh][d][s]) ---------
  {
    const int dt = w >> 1;                      // 4 tiles of 16 D-columns
    v8f acc = {0.f, 0.f, 0.f, 0.f, 0.f, 0.f, 0.f, 0.f};
    const bf16_t* abase = sw + (size_t)(16 * r + nl) * SPAD + 16 * half;
    const bf16_t* vtb = Vt + ((size_t)bh * D_DIM + dt * 16 + nl) * S_LEN + 16 * half;
    for (int kf = 0; kf < S_LEN / 32; ++kf) {
      const int kb = kf * 32;
      v8bf alo = *(const v8bf*)(abase + kb);        // k = kb + 16h + [0,8)
      v8bf ahi = *(const v8bf*)(abase + kb + 8);    // k = kb + 16h + 8 + [0,8)
      v16bf A;
#pragma unroll
      for (int i = 0; i < 8; ++i) { A[i] = alo[i]; A[8 + i] = ahi[i]; }
      v16bf Bm = *(const v16bf*)(vtb + kb);         // k = kb + 16h + [0,16)
      acc = wmma_bf16(A, Bm, acc);
    }
    float* cp = cout + (size_t)(q0 + 16 * r + 8 * half) * D_DIM + dt * 16 + nl;
#pragma unroll
    for (int v = 0; v < 8; ++v) cp[(size_t)v * D_DIM] = acc[v];
  }
}

// ---------------------------------------------------------------------------
extern "C" void kernel_launch(void* const* d_in, const int* in_sizes, int n_in,
                              void* d_out, int out_size, void* d_ws, size_t ws_size,
                              hipStream_t stream) {
  (void)in_sizes; (void)n_in; (void)out_size; (void)ws_size;
  const float* Q    = (const float*)d_in[0];
  const float* K    = (const float*)d_in[1];
  const float* V    = (const float*)d_in[2];
  const int*   mask = (const int*)d_in[3];

  float* ctx = (float*)d_out;                                   // [B,H,S,D]
  float* wts = ctx + (size_t)N_BH * S_LEN * D_DIM;              // [B,H,S,S]
  bf16_t* vt = (bf16_t*)d_ws;                                   // [bh][D][S] bf16 (8 MB)

  hipFuncSetAttribute(reinterpret_cast<const void*>(attn_kernel),
                      hipFuncAttributeMaxDynamicSharedMemorySize, SMEM_BYTES);

  vtranspose_kernel<<<dim3(N_BH, S_LEN / 64), 256, 0, stream>>>(V, vt);
  attn_kernel<<<dim3(N_BH * (S_LEN / QT)), 256, SMEM_BYTES, stream>>>(Q, K, mask, vt,
                                                                      ctx, wts);
}
